// AttentionResidualBlock_10299331576499
// MI455X (gfx1250) — matrix operand order repaired and checked by reading, not compiled, not run on hardware
//
#include <hip/hip_runtime.h>
#include <hip/hip_bf16.h>
#include <stdint.h>
#include <stddef.h>

#define BATCH 16
#define CH    256
#define HH    32
#define WWID  32
#define NSP   1024      /* HH*WWID */
#define NHEAD 8
#define HDIM  32
#define EPSV  1e-5f
#define SM_SCALE 0.17677669529663687f   /* 1/sqrt(32) */

typedef __attribute__((ext_vector_type(8)))  __bf16 v8bf;
typedef __attribute__((ext_vector_type(16))) __bf16 v16bf;
typedef __attribute__((ext_vector_type(8)))  float  v8f;

// ---------------------------------------------------------------- helpers
__device__ __forceinline__ v8f vzero8() {
  v8f r;
#pragma unroll
  for (int i = 0; i < 8; ++i) r[i] = 0.0f;
  return r;
}

// A-operand (16Mx32K bf16). Lane L holds row M = L%16.
// Order used here: lanes 0-15 hold K 0..15; lanes 16-31 K 16..31 (contiguous, matches B).
// rowk points at the start of this lane's 32-wide K segment.
__device__ __forceinline__ v16bf load_opA(const __bf16* rowk, int half) {
  v8bf lo = *(const v8bf*)(rowk + half * 16);
  v8bf hi = *(const v8bf*)(rowk + half * 16 + 8);
  v16bf r;
#pragma unroll
  for (int i = 0; i < 8; ++i) { r[i] = lo[i]; r[i + 8] = hi[i]; }
  return r;
}
// B-operand (32Kx16N bf16). Lane L holds column N = L%16.
// lanes 0-15: K 0..15 ; lanes 16-31: K 16..31 (contiguous per lane).
__device__ __forceinline__ v16bf load_opB(const __bf16* rowk, int half) {
  return *(const v16bf*)(rowk + half * 16);
}

__device__ __forceinline__ v8f wmma_bf16(v16bf a, v16bf b, v8f c) {
  return __builtin_amdgcn_wmma_f32_16x16x32_bf16(false, a, false, b, (short)0, c,
                                                 false, false);
}

__device__ __forceinline__ float half16_max(float v) {
#pragma unroll
  for (int m = 1; m < 16; m <<= 1) v = fmaxf(v, __shfl_xor(v, m, 32));
  return v;
}
__device__ __forceinline__ float half16_sum(float v) {
#pragma unroll
  for (int m = 1; m < 16; m <<= 1) v += __shfl_xor(v, m, 32);
  return v;
}

// Per-lane async global->LDS 16-byte copy (CDNA5 TDM-lite path, ASYNCcnt).
// Generic LDS addresses carry the group-relative offset in addr[31:0].
__device__ __forceinline__ void async_copy_b128(const __bf16* g, __bf16* l) {
  uint32_t loff = (uint32_t)(uintptr_t)l;
  asm volatile("global_load_async_to_lds_b128 %0, %1, off"
               :: "v"(loff), "v"(g) : "memory");
}
__device__ __forceinline__ void wait_asynccnt0() {
  asm volatile("s_wait_asynccnt 0x0" ::: "memory");
}

// ---------------------------------------------------------------- prep
// Fold BN scale into conv weight, re-layout [co][(kh*3+kw)*CH + ci], cvt bf16.
__global__ void arb_fold_conv_w(const float* __restrict__ w,
                                const float* __restrict__ gamma,
                                const float* __restrict__ var,
                                __bf16* __restrict__ wout) {
  int idx = blockIdx.x * blockDim.x + threadIdx.x;
  if (idx >= CH * CH * 9) return;
  int co = idx / (CH * 9);
  int rem = idx - co * (CH * 9);
  int ci = rem / 9;
  int kk = rem - ci * 9;
  float inv = gamma[co] * rsqrtf(var[co] + EPSV);
  wout[(size_t)co * (9 * CH) + (size_t)kk * CH + ci] = (__bf16)(w[idx] * inv);
}

__global__ void arb_fold_conv_b(const float* __restrict__ cb,
                                const float* __restrict__ gamma,
                                const float* __restrict__ beta,
                                const float* __restrict__ mean,
                                const float* __restrict__ var,
                                float* __restrict__ bout) {
  int c = threadIdx.x;
  if (c >= CH) return;
  float inv = gamma[c] * rsqrtf(var[c] + EPSV);
  bout[c] = cb[c] * inv + beta[c] - mean[c] * inv;
}

__global__ void arb_cvt_w(const float* __restrict__ w, __bf16* __restrict__ wout,
                          int n) {
  int idx = blockIdx.x * blockDim.x + threadIdx.x;
  if (idx < n) wout[idx] = (__bf16)w[idx];
}

// x [b][c][n] f32  ->  XT [b][n][c] bf16
__global__ void arb_transpose_x(const float* __restrict__ x,
                                __bf16* __restrict__ XT) {
  int idx = blockIdx.x * blockDim.x + threadIdx.x;
  if (idx >= BATCH * NSP * CH) return;
  int b = idx / (NSP * CH);
  int rem = idx - b * (NSP * CH);
  int n = rem / CH;
  int c = rem - n * CH;
  XT[idx] = (__bf16)x[(size_t)b * CH * NSP + (size_t)c * NSP + n];
}

// ---------------------------------------------------------------- fused q/k/v
// q/k/v[co,n] = sum_c W[co][c] * XT[b][n][c] (+bias). One B-tile feeds 3 WMMAs.
__global__ __launch_bounds__(128) void arb_qkv(const __bf16* __restrict__ XT,
                                               const __bf16* __restrict__ WQ,
                                               const __bf16* __restrict__ WK,
                                               const __bf16* __restrict__ WV,
                                               const float* __restrict__ qb,
                                               const float* __restrict__ kb,
                                               const float* __restrict__ vb,
                                               __bf16* __restrict__ QT,
                                               __bf16* __restrict__ KT,
                                               __bf16* __restrict__ VC) {
  int lane = threadIdx.x & 31, wid = threadIdx.x >> 5;
  int half = lane >> 4, l15 = lane & 15;
  int t = blockIdx.x * 4 + wid;              // 16384 tiles
  int b = t / 1024;                          // 64 n-tiles * 16 co-tiles
  int rem = t - b * 1024;
  int n0 = (rem >> 4) * 16, co0 = (rem & 15) * 16;
  const __bf16* brow = XT + (size_t)b * NSP * CH + (size_t)(n0 + l15) * CH;
  const __bf16* aq = WQ + (size_t)(co0 + l15) * CH;
  const __bf16* ak = WK + (size_t)(co0 + l15) * CH;
  const __bf16* av = WV + (size_t)(co0 + l15) * CH;
  v8f accq = vzero8(), acck = vzero8(), accv = vzero8();
#pragma unroll
  for (int c0 = 0; c0 < CH; c0 += 32) {
    __builtin_prefetch(brow + c0 + 64, 0, 1);
    v16bf bm = load_opB(brow + c0, half);
    accq = wmma_bf16(load_opA(aq + c0, half), bm, accq);
    acck = wmma_bf16(load_opA(ak + c0, half), bm, acck);
    accv = wmma_bf16(load_opA(av + c0, half), bm, accv);
  }
#pragma unroll
  for (int r = 0; r < 8; ++r) {
    int co = co0 + r + half * 8;
    int n = n0 + l15;
    size_t nc = (size_t)b * NSP * CH + (size_t)n * CH + co;
    QT[nc] = (__bf16)(accq[r] + qb[co]);
    KT[nc] = (__bf16)(acck[r] + kb[co]);
    VC[(size_t)b * CH * NSP + (size_t)co * NSP + n] = (__bf16)(accv[r] + vb[co]);
  }
}

// ---------------------------------------------------------------- conv3x3 (implicit GEMM)
// Block = 4 waves = 64 co x 16 n tile. The 3x18x256 bf16 activation halo tile
// (incl. zero padding) is staged once per block into LDS with per-lane async
// global->LDS B128 copies (ASYNCcnt), then each wave runs 72 WMMAs with the
// B-operand read from LDS and the folded weights streamed from L2.
// MODE 1: bias + relu -> bf16 [b][n][co]  (feeds conv2)
// MODE 2: bias        -> f32  [b][co][n]  (conv branch output)
template <int MODE>
__global__ __launch_bounds__(128) void arb_conv3x3(const __bf16* __restrict__ SRC,
                                                   const __bf16* __restrict__ WC,
                                                   const float* __restrict__ BIAS,
                                                   void* __restrict__ OUTP) {
  __shared__ __align__(64) __bf16 lbuf[3][18][CH];   // 27648 B
  int lane = threadIdx.x & 31, wid = threadIdx.x >> 5;
  int half = lane >> 4, l15 = lane & 15;
  int t = blockIdx.x;                       // BATCH * 4 co-blocks * 64 n-tiles
  int b = t >> 8;
  int rem = t & 255;
  int cb = rem >> 6;                        // co-block of 64
  int nt = rem & 63;                        // 16-wide spatial tile
  int co_base = cb * 64;
  int n0 = nt * 16;
  int h = n0 >> 5, w0 = n0 & 31;
  const __bf16* sb = SRC + (size_t)b * NSP * CH;

  // ---- stage halo tile: 3 rows x 18 cols x 256 ch, 16B granules
  for (int idx = threadIdx.x; idx < 3 * 18 * 32; idx += 128) {
    int slot = idx >> 5;                    // 0..53 spatial slot
    int chunk = idx & 31;                   // 8 bf16 each
    int kh = slot / 18;
    int pos = slot - kh * 18;
    int hp = h + kh - 1;
    int wq = w0 + pos - 1;
    __bf16* dst = &lbuf[kh][pos][chunk * 8];
    if ((unsigned)hp < (unsigned)HH && (unsigned)wq < (unsigned)WWID) {
      async_copy_b128(sb + (size_t)(hp * WWID + wq) * CH + chunk * 8, dst);
    } else {
      *(int4*)dst = make_int4(0, 0, 0, 0);  // zero padding
    }
  }
  wait_asynccnt0();
  __syncthreads();

  // ---- 72 WMMAs per wave, B from LDS, A from global (L2-resident weights)
  const __bf16* arow = WC + (size_t)(co_base + wid * 16 + l15) * (9 * CH);
  v8f acc = vzero8();
#pragma unroll
  for (int kh = 0; kh < 3; ++kh) {
#pragma unroll
    for (int kw = 0; kw < 3; ++kw) {
      const __bf16* lrow = &lbuf[kh][l15 + kw][0];
      int kbase = (kh * 3 + kw) * CH;
#pragma unroll
      for (int c0 = 0; c0 < CH; c0 += 32) {
        v16bf bm = load_opB(lrow + c0, half);
        v16bf am = load_opA(arow + kbase + c0, half);
        acc = wmma_bf16(am, bm, acc);
      }
    }
  }
#pragma unroll
  for (int r = 0; r < 8; ++r) {
    int co = co_base + wid * 16 + r + half * 8;
    int n = n0 + l15;
    float v = acc[r] + BIAS[co];
    if (MODE == 1) {
      v = fmaxf(v, 0.0f);
      ((__bf16*)OUTP)[(size_t)b * NSP * CH + (size_t)n * CH + co] = (__bf16)v;
    } else {
      ((float*)OUTP)[(size_t)b * CH * NSP + (size_t)co * NSP + n] = v;
    }
  }
}

// ---------------------------------------------------------------- flash attention
// Per wave: one (b, head, 16-query) tile. Online softmax, O accumulated in f32.
__global__ __launch_bounds__(128) void arb_flash(const __bf16* __restrict__ QT,
                                                 const __bf16* __restrict__ KT,
                                                 const __bf16* __restrict__ VC,
                                                 __bf16* __restrict__ AT) {
  __shared__ __align__(64) __bf16 plds[4][16][32];
  int lane = threadIdx.x & 31, wid = threadIdx.x >> 5;
  int half = lane >> 4, l15 = lane & 15;
  int t = blockIdx.x * 4 + wid;          // 8192 tiles
  int b = t >> 9;                        // NHEAD*64 = 512 per batch
  int rem = t & 511;
  int hd = rem >> 6;
  int n0 = (rem & 63) * 16;
  int hbase = hd * HDIM;
  const __bf16* qb = QT + (size_t)b * NSP * CH;
  const __bf16* kb = KT + (size_t)b * NSP * CH;
  const __bf16* vb = VC + (size_t)b * CH * NSP;

  v16bf aq = load_opA(qb + (size_t)(n0 + l15) * CH + hbase, half); // 16q x 32d
  v8f o0 = vzero8(), o1 = vzero8();
  float rmax[8], rsum[8];
#pragma unroll
  for (int r = 0; r < 8; ++r) { rmax[r] = -3.0e38f; rsum[r] = 0.0f; }

  for (int jb = 0; jb < 32; ++jb) {      // 32 blocks of 32 keys
    int key0 = jb * 32;
    v16bf bk0 = load_opB(kb + (size_t)(key0 + l15) * CH + hbase, half);
    v16bf bk1 = load_opB(kb + (size_t)(key0 + 16 + l15) * CH + hbase, half);
    v8f s0 = wmma_bf16(aq, bk0, vzero8());   // K = head_dim = 32 exactly
    v8f s1 = wmma_bf16(aq, bk1, vzero8());
#pragma unroll
    for (int r = 0; r < 8; ++r) {
      float a0 = s0[r] * SM_SCALE, a1 = s1[r] * SM_SCALE;
      float mt = half16_max(fmaxf(a0, a1));      // row lives in one 16-lane half
      float nm = fmaxf(rmax[r], mt);
      float corr = __expf(rmax[r] - nm);
      rmax[r] = nm;
      float p0 = __expf(a0 - nm), p1 = __expf(a1 - nm);
      rsum[r] = rsum[r] * corr + half16_sum(p0 + p1);
      o0[r] *= corr;
      o1[r] *= corr;
      int row = r + half * 8;
      plds[wid][row][l15] = (__bf16)p0;          // P in [query][key] order
      plds[wid][row][16 + l15] = (__bf16)p1;
    }
    __syncthreads();
    v16bf ap = load_opA(&plds[wid][l15][0], half);        // 16q x 32keys
    const __bf16* vrow = vb + (size_t)(hbase + l15) * NSP + key0;
    v16bf bv0 = load_opB(vrow, half);                     // d 0..15
    v16bf bv1 = load_opB(vrow + (size_t)16 * NSP, half);  // d 16..31
    o0 = wmma_bf16(ap, bv0, o0);
    o1 = wmma_bf16(ap, bv1, o1);
    __syncthreads();
  }
#pragma unroll
  for (int r = 0; r < 8; ++r) {
    float inv = 1.0f / rsum[r];
    int qrow = n0 + r + half * 8;
    size_t base = (size_t)b * NSP * CH + (size_t)qrow * CH + hbase;
    AT[base + l15] = (__bf16)(o0[r] * inv);
    AT[base + 16 + l15] = (__bf16)(o1[r] * inv);
  }
}

// ---------------------------------------------------------------- out projection
__global__ __launch_bounds__(128) void arb_proj(const __bf16* __restrict__ AT,
                                                const __bf16* __restrict__ WO,
                                                const float* __restrict__ ob,
                                                float* __restrict__ AO) {
  int lane = threadIdx.x & 31, wid = threadIdx.x >> 5;
  int half = lane >> 4, l15 = lane & 15;
  int t = blockIdx.x * 4 + wid;
  int b = t / 1024;
  int rem = t - b * 1024;
  int n0 = (rem >> 4) * 16, co0 = (rem & 15) * 16;
  const __bf16* brow = AT + (size_t)b * NSP * CH + (size_t)(n0 + l15) * CH;
  const __bf16* arow = WO + (size_t)(co0 + l15) * CH;
  v8f acc = vzero8();
#pragma unroll
  for (int c0 = 0; c0 < CH; c0 += 32)
    acc = wmma_bf16(load_opA(arow + c0, half), load_opB(brow + c0, half), acc);
#pragma unroll
  for (int r = 0; r < 8; ++r) {
    int co = co0 + r + half * 8;
    int n = n0 + l15;
    AO[(size_t)b * CH * NSP + (size_t)co * NSP + n] = acc[r] + ob[co];
  }
}

// ---------------------------------------------------------------- final fuse
__global__ void arb_fuse(const float* __restrict__ C2, const float* __restrict__ AO,
                         const float* __restrict__ x, const float* __restrict__ gate,
                         float* __restrict__ out) {
  int idx = blockIdx.x * blockDim.x + threadIdx.x;
  if (idx >= BATCH * CH * NSP) return;
  float g = 1.0f / (1.0f + __expf(-gate[0]));
  out[idx] = fmaxf(0.0f, C2[idx] + g * AO[idx] + x[idx]);
}

// ---------------------------------------------------------------- launch
extern "C" void kernel_launch(void* const* d_in, const int* in_sizes, int n_in,
                              void* d_out, int out_size, void* d_ws, size_t ws_size,
                              hipStream_t stream) {
  (void)in_sizes; (void)n_in; (void)out_size; (void)ws_size;
  const float* x        = (const float*)d_in[0];
  const float* conv1_w  = (const float*)d_in[1];
  const float* conv1_b  = (const float*)d_in[2];
  const float* bn1_g    = (const float*)d_in[3];
  const float* bn1_be   = (const float*)d_in[4];
  const float* bn1_m    = (const float*)d_in[5];
  const float* bn1_v    = (const float*)d_in[6];
  const float* conv2_w  = (const float*)d_in[7];
  const float* conv2_b  = (const float*)d_in[8];
  const float* bn2_g    = (const float*)d_in[9];
  const float* bn2_be   = (const float*)d_in[10];
  const float* bn2_m    = (const float*)d_in[11];
  const float* bn2_v    = (const float*)d_in[12];
  const float* q_w      = (const float*)d_in[13];
  const float* q_b      = (const float*)d_in[14];
  const float* k_w      = (const float*)d_in[15];
  const float* k_b      = (const float*)d_in[16];
  const float* v_w      = (const float*)d_in[17];
  const float* v_b      = (const float*)d_in[18];
  const float* out_w    = (const float*)d_in[19];
  const float* out_b    = (const float*)d_in[20];
  const float* gate     = (const float*)d_in[21];
  float* out = (float*)d_out;

  // workspace partition (256B aligned slices)
  char* w = (char*)d_ws;
  size_t off = 0;
  auto take = [&](size_t bytes) -> void* {
    void* p = w + off;
    off += (bytes + 255) & ~(size_t)255;
    return p;
  };
  const size_t BNC2 = (size_t)BATCH * NSP * CH * 2;   // 8 MiB bf16
  const size_t BNC4 = (size_t)BATCH * NSP * CH * 4;   // 16 MiB f32
  __bf16* XT = (__bf16*)take(BNC2);
  __bf16* Y1 = (__bf16*)take(BNC2);
  __bf16* QT = (__bf16*)take(BNC2);
  __bf16* KT = (__bf16*)take(BNC2);
  __bf16* VC = (__bf16*)take(BNC2);
  __bf16* AT = (__bf16*)take(BNC2);
  float*  C2 = (float*)take(BNC4);
  float*  AO = (float*)take(BNC4);
  __bf16* W1 = (__bf16*)take((size_t)CH * 9 * CH * 2);
  __bf16* W2 = (__bf16*)take((size_t)CH * 9 * CH * 2);
  __bf16* WQ = (__bf16*)take((size_t)CH * CH * 2);
  __bf16* WK = (__bf16*)take((size_t)CH * CH * 2);
  __bf16* WV = (__bf16*)take((size_t)CH * CH * 2);
  __bf16* WO = (__bf16*)take((size_t)CH * CH * 2);
  float*  B1 = (float*)take(CH * 4);
  float*  B2 = (float*)take(CH * 4);

  // ---- weight prep (BN folding + bf16 conversion)
  arb_fold_conv_w<<<2304, 256, 0, stream>>>(conv1_w, bn1_g, bn1_v, W1);
  arb_fold_conv_w<<<2304, 256, 0, stream>>>(conv2_w, bn2_g, bn2_v, W2);
  arb_fold_conv_b<<<1, 256, 0, stream>>>(conv1_b, bn1_g, bn1_be, bn1_m, bn1_v, B1);
  arb_fold_conv_b<<<1, 256, 0, stream>>>(conv2_b, bn2_g, bn2_be, bn2_m, bn2_v, B2);
  arb_cvt_w<<<256, 256, 0, stream>>>(q_w, WQ, CH * CH);
  arb_cvt_w<<<256, 256, 0, stream>>>(k_w, WK, CH * CH);
  arb_cvt_w<<<256, 256, 0, stream>>>(v_w, WV, CH * CH);
  arb_cvt_w<<<256, 256, 0, stream>>>(out_w, WO, CH * CH);

  // ---- activation prep
  arb_transpose_x<<<16384, 256, 0, stream>>>(x, XT);

  // ---- main pipeline
  arb_qkv<<<4096, 128, 0, stream>>>(XT, WQ, WK, WV, q_b, k_b, v_b, QT, KT, VC);
  arb_conv3x3<1><<<4096, 128, 0, stream>>>(XT, W1, B1, (void*)Y1);
  arb_flash<<<2048, 128, 0, stream>>>(QT, KT, VC, AT);
  arb_conv3x3<2><<<4096, 128, 0, stream>>>(Y1, W2, B2, (void*)C2);
  arb_proj<<<4096, 128, 0, stream>>>(AT, WO, out_b, AO);
  arb_fuse<<<16384, 256, 0, stream>>>(C2, AO, x, gate, out);
}
